// ConditionedLatentSDETransition_30305289240756
// MI455X (gfx1250) — matrix operand order repaired and checked by reading, not compiled, not run on hardware
//
#include <hip/hip_runtime.h>
#include <hip/hip_bf16.h>
#include <math.h>

// ---------------------------------------------------------------------------
// Types for CDNA5 WMMA (wave32, v_wmma_f32_16x16x32_bf16)
// ---------------------------------------------------------------------------
typedef __attribute__((ext_vector_type(16))) __bf16 v16bf;
typedef __attribute__((ext_vector_type(8)))  float  v8f;

union BFrag { uint4 u[2]; v16bf v; };

// ---------------------------------------------------------------------------
// CDNA5 async copy helpers: global -> LDS direct, tracked by ASYNCcnt.
// ---------------------------------------------------------------------------
__device__ __forceinline__ void async_ld_b128(uint32_t lds_addr, const void* gaddr)
{
    asm volatile("global_load_async_to_lds_b128 %0, %1, off"
                 :: "v"(lds_addr), "v"(gaddr) : "memory");
}

__device__ __forceinline__ void wait_async0()
{
#if __has_builtin(__builtin_amdgcn_s_wait_asynccnt)
    __builtin_amdgcn_s_wait_asynccnt(0);
#else
    asm volatile("s_wait_asynccnt 0x0" ::: "memory");
#endif
}

__device__ __forceinline__ void wait_async4()
{
#if __has_builtin(__builtin_amdgcn_s_wait_asynccnt)
    __builtin_amdgcn_s_wait_asynccnt(4);
#else
    asm volatile("s_wait_asynccnt 0x4" ::: "memory");
#endif
}

// ---------------------------------------------------------------------------
// Tiled bf16 GEMM:  out[M x Nstore] = X[M x K] * W[Nw x K]^T (+bias)(+addsrc)
// Block tile 128x128, BK=32, 8 waves, each wave 32x64 via 2x4 wmma frags.
// Global->LDS staging uses GLOBAL_LOAD_ASYNC_TO_LDS_B128 with a double-
// buffered LDS pipeline: tile k+1 streams in while tile k runs WMMA.
// Weight rows are clamped to Nw-1 (garbage only reaches columns never
// stored: col < Nstore <= Nw). addsrc may alias out.
// ---------------------------------------------------------------------------
#define LSTR 40  // LDS row stride in bf16 elements (32 data + 8 pad), 80B

__global__ __launch_bounds__(256)
void gemm_bf16_kernel(const __bf16* __restrict__ X, const __bf16* __restrict__ W,
                      const float* __restrict__ bias, const float* __restrict__ addsrc,
                      float* __restrict__ out,
                      int M, int K, int Nw, int Nstore, int ldOut)
{
    __shared__ __bf16 As[2][128 * LSTR];
    __shared__ __bf16 Bs[2][128 * LSTR];

    const int tid   = threadIdx.x;
    const int lane  = tid & 31;
    const int wid   = tid >> 5;       // 0..7
    const int waveM = wid & 3;        // 4 row-slabs of 32
    const int waveN = wid >> 2;       // 2 col-slabs of 64
    const int bm    = blockIdx.y * 128;
    const int bn    = blockIdx.x * 128;

    const int h   = lane >> 4;        // half-wave selector
    const int l16 = lane & 15;

    // cooperative staging mapping: 4 threads per row, 8 bf16 (16B) each
    const int lr = tid >> 2;          // 0..63
    const int lc = (tid & 3) * 8;     // 0,8,16,24

    // per-thread global source pointers (advance by k in elements)
    const __bf16* aSrc0 = X + (size_t)(bm + lr) * K + lc;
    const __bf16* aSrc1 = aSrc0 + (size_t)64 * K;
    int nb0 = bn + lr;      if (nb0 > Nw - 1) nb0 = Nw - 1;
    int nb1 = bn + lr + 64; if (nb1 > Nw - 1) nb1 = Nw - 1;
    const __bf16* bSrc0 = W + (size_t)nb0 * K + lc;
    const __bf16* bSrc1 = W + (size_t)nb1 * K + lc;

    // LDS byte addresses (flat-pointer low 32 bits == LDS offset)
    uint32_t ldsA0[2], ldsA1[2], ldsB0[2], ldsB1[2];
#pragma unroll
    for (int p = 0; p < 2; ++p) {
        ldsA0[p] = (uint32_t)(uintptr_t)&As[p][lr * LSTR + lc];
        ldsA1[p] = (uint32_t)(uintptr_t)&As[p][(lr + 64) * LSTR + lc];
        ldsB0[p] = (uint32_t)(uintptr_t)&Bs[p][lr * LSTR + lc];
        ldsB1[p] = (uint32_t)(uintptr_t)&Bs[p][(lr + 64) * LSTR + lc];
    }

    v8f acc[2][4];
#pragma unroll
    for (int mi = 0; mi < 2; ++mi)
#pragma unroll
        for (int ni = 0; ni < 4; ++ni)
            acc[mi][ni] = (v8f){0.f,0.f,0.f,0.f,0.f,0.f,0.f,0.f};

    const int nk = K >> 5;

    // prologue: stream tile 0 into buffer 0
    async_ld_b128(ldsA0[0], aSrc0);
    async_ld_b128(ldsA1[0], aSrc1);
    async_ld_b128(ldsB0[0], bSrc0);
    async_ld_b128(ldsB1[0], bSrc1);

    for (int i = 0; i < nk; ++i) {
        const int p = i & 1;
        if (i + 1 < nk) {
            const int k = (i + 1) << 5;
            async_ld_b128(ldsA0[p ^ 1], aSrc0 + k);
            async_ld_b128(ldsA1[p ^ 1], aSrc1 + k);
            async_ld_b128(ldsB0[p ^ 1], bSrc0 + k);
            async_ld_b128(ldsB1[p ^ 1], bSrc1 + k);
            wait_async4();   // loads retire in order: <=4 left => tile i landed
        } else {
            wait_async0();
        }
        __syncthreads();

        // B fragments: lane = col (l16), contiguous K range [16h, 16h+16)
        BFrag bf[4];
#pragma unroll
        for (int ni = 0; ni < 4; ++ni) {
            const __bf16* pB = &Bs[p][(waveN * 64 + ni * 16 + l16) * LSTR + 16 * h];
            bf[ni].u[0] = *(const uint4*)pB;
            bf[ni].u[1] = *(const uint4*)(pB + 8);
        }
        // A fragments + 8 WMMAs
#pragma unroll
        for (int mi = 0; mi < 2; ++mi) {
            const __bf16* pA = &As[p][(waveM * 32 + mi * 16 + l16) * LSTR];
            BFrag af;
            af.u[0] = *(const uint4*)(pA + 16 * h);       // K [16h, 16h+8)
            af.u[1] = *(const uint4*)(pA + 16 * h + 8);   // K [16h+8, 16h+16)
#pragma unroll
            for (int ni = 0; ni < 4; ++ni)
                acc[mi][ni] = __builtin_amdgcn_wmma_f32_16x16x32_bf16(
                    false, af.v, false, bf[ni].v, (short)0, acc[mi][ni], false, false);
        }
        __syncthreads();   // protect buffer p before iteration i+1 overwrites it
    }

    // epilogue: C layout -> lane holds col l16, rows mi*16 + 8h + r
#pragma unroll
    for (int mi = 0; mi < 2; ++mi)
#pragma unroll
        for (int ni = 0; ni < 4; ++ni) {
            int col = bn + waveN * 64 + ni * 16 + l16;
            if (col < Nstore) {
#pragma unroll
                for (int r = 0; r < 8; ++r) {
                    int row = bm + waveM * 32 + mi * 16 + 8 * h + r;
                    float v = acc[mi][ni][r];
                    if (bias)   v += bias[col];
                    size_t idx = (size_t)row * ldOut + col;
                    if (addsrc) v += addsrc[idx];
                    out[idx] = v;
                }
            }
        }
}

// ---------------------------------------------------------------------------
// Column statistics over batch axis (deterministic split reduction)
// ---------------------------------------------------------------------------
__global__ __launch_bounds__(256)
void k_colstat_partial(const float* __restrict__ P, float* __restrict__ ps,
                       float* __restrict__ pq, int N, int rowsPer)
{
    int col   = blockIdx.x * 64 + (threadIdx.x & 63);
    int rlane = threadIdx.x >> 6;
    int r0    = blockIdx.y * rowsPer;
    float s = 0.f, q = 0.f;
    for (int r = r0 + rlane; r < r0 + rowsPer; r += 4) {
        float v = P[(size_t)r * N + col];
        s += v; q += v * v;
    }
    __shared__ float sh[512];
    sh[threadIdx.x] = s; sh[256 + threadIdx.x] = q;
    __syncthreads();
    if (threadIdx.x < 64) {
        float a = sh[threadIdx.x] + sh[threadIdx.x + 64] + sh[threadIdx.x + 128] + sh[threadIdx.x + 192];
        float b = sh[256 + threadIdx.x] + sh[256 + threadIdx.x + 64] +
                  sh[256 + threadIdx.x + 128] + sh[256 + threadIdx.x + 192];
        ps[(size_t)blockIdx.y * N + col] = a;
        pq[(size_t)blockIdx.y * N + col] = b;
    }
}

__global__ void k_colstat_final(const float* __restrict__ ps, const float* __restrict__ pq,
                                float* __restrict__ mu, float* __restrict__ rinv,
                                int N, int parts, float invM)
{
    int c = blockIdx.x * 256 + threadIdx.x;
    if (c < N) {
        float s = 0.f, q = 0.f;
        for (int i = 0; i < parts; ++i) { s += ps[(size_t)i * N + c]; q += pq[(size_t)i * N + c]; }
        float m   = s * invM;
        float var = q * invM - m * m;
        mu[c]   = m;
        rinv[c] = rsqrtf(var + 1e-5f);
    }
}

// ---------------------------------------------------------------------------
// Elementwise kernels
// ---------------------------------------------------------------------------
__global__ void k_bnrelu(const float* __restrict__ P, const float* __restrict__ mu,
                         const float* __restrict__ rinv, const float* __restrict__ g,
                         const float* __restrict__ be, __bf16* __restrict__ A,
                         int nmask, int total)
{
    int i = blockIdx.x * 256 + threadIdx.x;
    if (i < total) {
        int c = i & nmask;
        float v = (P[i] - mu[c]) * rinv[c] * g[c] + be[c];
        A[i] = (__bf16)(v > 0.f ? v : 0.f);
    }
}

__global__ void k_tanh(const float* __restrict__ P, __bf16* __restrict__ T, int total)
{
    int i = blockIdx.x * 256 + threadIdx.x;
    if (i < total) T[i] = (__bf16)tanhf(P[i]);
}

__global__ void k_softplus(float* __restrict__ G, int total)
{
    int i = blockIdx.x * 256 + threadIdx.x;
    if (i < total) {
        float x = G[i];
        float r = (x > 20.f) ? x : log1pf(expf(x));
        G[i] = r + 1e-5f;
    }
}

__global__ void k_init_z(const float* __restrict__ zd, float* __restrict__ z,
                         __bf16* __restrict__ zb, int total)
{
    int i = blockIdx.x * 256 + threadIdx.x;
    if (i < total) { float v = zd[i]; z[i] = v; zb[i] = (__bf16)v; }
}

__global__ void k_cvt_w(const float* __restrict__ src, __bf16* __restrict__ dst,
                        int rowsSrc, int K, int total)
{
    int i = blockIdx.x * 256 + threadIdx.x;
    if (i < total) {
        int r = i / K;
        dst[i] = (r < rowsSrc) ? (__bf16)src[i] : (__bf16)0.f;
    }
}

__global__ void k_euler(float* __restrict__ z, __bf16* __restrict__ zb,
                        const float* __restrict__ F, const float* __restrict__ G,
                        const float* __restrict__ E, const float* __restrict__ dtp, int total)
{
    int i = blockIdx.x * 256 + threadIdx.x;
    if (i < total) {
        float dt = *dtp;
        float hh = dt * 0.125f;
        float sh = sqrtf(fabsf(hh) + 1e-8f);
        float v = z[i] + hh * F[i] + sh * G[i] * E[i];
        z[i] = v; zb[i] = (__bf16)v;
    }
}

__global__ void k_utdt(const float* __restrict__ ut, __bf16* __restrict__ ub,
                       const float* __restrict__ dtp, int total)
{
    int i = blockIdx.x * 256 + threadIdx.x;
    if (i < total) { float dt = *dtp; ub[i] = (__bf16)(ut[i] * dt); }
}

__global__ void k_cvt_f2b(const float* __restrict__ s, __bf16* __restrict__ d, int total)
{
    int i = blockIdx.x * 256 + threadIdx.x;
    if (i < total) d[i] = (__bf16)s[i];
}

// ---------------------------------------------------------------------------
// Host-side orchestration
// ---------------------------------------------------------------------------
extern "C" void kernel_launch(void* const* d_in, const int* in_sizes, int n_in,
                              void* d_out, int out_size, void* d_ws, size_t ws_size,
                              hipStream_t stream)
{
    constexpr int Bz = 4096, Dd = 512, H1 = 1024, H2 = 1024, HD = 512;
    constexpr int Uu = 64, NOBS = 25, NPAD = 32, NE = 8;

    const float* z_dyn = (const float*)d_in[0];
    const float* dtp   = (const float*)d_in[2];
    const float* ut    = (const float*)d_in[3];
    const float* eps   = (const float*)d_in[4];
    const float* W1    = (const float*)d_in[5];
    const float* b1    = (const float*)d_in[6];
    const float* g1    = (const float*)d_in[7];
    const float* be1   = (const float*)d_in[8];
    const float* W2    = (const float*)d_in[9];
    const float* b2    = (const float*)d_in[10];
    const float* g2    = (const float*)d_in[11];
    const float* be2   = (const float*)d_in[12];
    const float* W3    = (const float*)d_in[13];
    const float* b3    = (const float*)d_in[14];
    const float* Wd1   = (const float*)d_in[15];
    const float* bd1   = (const float*)d_in[16];
    const float* Wd2   = (const float*)d_in[17];
    const float* bd2   = (const float*)d_in[18];
    const float* Bsde  = (const float*)d_in[19];
    const float* Cm    = (const float*)d_in[20];
    const float* Dm    = (const float*)d_in[21];
    float* out = (float*)d_out;

    // workspace carve-out (~72 MB total)
    char* cur = (char*)d_ws;
    auto alloc = [&](size_t bytes) -> char* {
        char* p = cur; cur += (bytes + 255) & ~(size_t)255; return p;
    };
    float*  zcur = (float*)alloc((size_t)Bz * Dd * 4);
    __bf16* zb   = (__bf16*)alloc((size_t)Bz * Dd * 2);
    float*  P    = (float*)alloc((size_t)Bz * H1 * 4);   // pre-activation scratch
    __bf16* A1   = (__bf16*)alloc((size_t)Bz * H1 * 2);
    __bf16* A2   = (__bf16*)alloc((size_t)Bz * H2 * 2);
    float*  Fb   = (float*)alloc((size_t)Bz * Dd * 4);   // drift
    __bf16* Tb   = (__bf16*)alloc((size_t)Bz * HD * 2);  // tanh intermediate
    float*  Gb   = (float*)alloc((size_t)Bz * Dd * 4);   // diffusion
    float*  mu   = (float*)alloc(H1 * 4);
    float*  rinv = (float*)alloc(H1 * 4);
    float*  ps   = (float*)alloc((size_t)32 * H1 * 4);
    float*  pq   = (float*)alloc((size_t)32 * H1 * 4);
    __bf16* utb  = (__bf16*)alloc((size_t)Bz * Uu * 2);
    __bf16* W1b  = (__bf16*)alloc((size_t)H1 * Dd * 2);
    __bf16* W2b  = (__bf16*)alloc((size_t)H2 * H1 * 2);
    __bf16* W3b  = (__bf16*)alloc((size_t)Dd * H2 * 2);
    __bf16* Wd1b = (__bf16*)alloc((size_t)HD * Dd * 2);
    __bf16* Wd2b = (__bf16*)alloc((size_t)Dd * HD * 2);
    __bf16* Bsb  = (__bf16*)alloc((size_t)Dd * Uu * 2);
    __bf16* Cb   = (__bf16*)alloc((size_t)NPAD * Dd * 2);
    __bf16* Db   = (__bf16*)alloc((size_t)NPAD * Uu * 2);
    (void)ws_size; (void)in_sizes; (void)n_in; (void)out_size;

    auto ew = [&](int total) { return dim3((total + 255) / 256); };
    auto gemm = [&](const __bf16* X, const __bf16* Wt, const float* bias,
                    const float* addsrc, float* o, int K, int Nw, int Nstore,
                    int ldOut, int ntiles) {
        dim3 grid(ntiles, Bz / 128);
        gemm_bf16_kernel<<<grid, 256, 0, stream>>>(X, Wt, bias, addsrc, o,
                                                   Bz, K, Nw, Nstore, ldOut);
    };

    // weight conversion to bf16 (zero-padded rows for C, D)
    k_cvt_w<<<ew(H1 * Dd),   256, 0, stream>>>(W1,   W1b,  H1,   Dd, H1 * Dd);
    k_cvt_w<<<ew(H2 * H1),   256, 0, stream>>>(W2,   W2b,  H2,   H1, H2 * H1);
    k_cvt_w<<<ew(Dd * H2),   256, 0, stream>>>(W3,   W3b,  Dd,   H2, Dd * H2);
    k_cvt_w<<<ew(HD * Dd),   256, 0, stream>>>(Wd1,  Wd1b, HD,   Dd, HD * Dd);
    k_cvt_w<<<ew(Dd * HD),   256, 0, stream>>>(Wd2,  Wd2b, Dd,   HD, Dd * HD);
    k_cvt_w<<<ew(Dd * Uu),   256, 0, stream>>>(Bsde, Bsb,  Dd,   Uu, Dd * Uu);
    k_cvt_w<<<ew(NPAD * Dd), 256, 0, stream>>>(Cm,   Cb,   NOBS, Dd, NPAD * Dd);
    k_cvt_w<<<ew(NPAD * Uu), 256, 0, stream>>>(Dm,   Db,   NOBS, Uu, NPAD * Uu);
    k_init_z<<<ew(Bz * Dd), 256, 0, stream>>>(z_dyn, zcur, zb, Bz * Dd);

    for (int s = 0; s < NE; ++s) {
        // ---- drift ----
        gemm(zb, W1b, b1, nullptr, P, Dd, H1, H1, H1, 8);
        k_colstat_partial<<<dim3(H1 / 64, 32), 256, 0, stream>>>(P, ps, pq, H1, 128);
        k_colstat_final<<<ew(H1), 256, 0, stream>>>(ps, pq, mu, rinv, H1, 32, 1.f / Bz);
        k_bnrelu<<<ew(Bz * H1), 256, 0, stream>>>(P, mu, rinv, g1, be1, A1, H1 - 1, Bz * H1);

        gemm(A1, W2b, b2, nullptr, P, H1, H2, H2, H2, 8);
        k_colstat_partial<<<dim3(H2 / 64, 32), 256, 0, stream>>>(P, ps, pq, H2, 128);
        k_colstat_final<<<ew(H2), 256, 0, stream>>>(ps, pq, mu, rinv, H2, 32, 1.f / Bz);
        k_bnrelu<<<ew(Bz * H2), 256, 0, stream>>>(P, mu, rinv, g2, be2, A2, H2 - 1, Bz * H2);

        gemm(A2, W3b, b3, nullptr, Fb, H2, Dd, Dd, Dd, 4);

        // ---- diffusion (from step-start z) ----
        gemm(zb, Wd1b, bd1, nullptr, P, Dd, HD, HD, HD, 4);
        k_tanh<<<ew(Bz * HD), 256, 0, stream>>>(P, Tb, Bz * HD);
        gemm(Tb, Wd2b, bd2, nullptr, Gb, HD, Dd, Dd, Dd, 4);
        k_softplus<<<ew(Bz * Dd), 256, 0, stream>>>(Gb, Bz * Dd);

        // ---- Euler-Maruyama update ----
        k_euler<<<ew(Bz * Dd), 256, 0, stream>>>(zcur, zb, Fb, Gb,
                                                 eps + (size_t)s * Bz * Dd, dtp, Bz * Dd);
    }

    // ---- outputs: z_next = z + (ut*dt)@B_sde^T ; yt = z_next@C^T + (ut*dt)@D^T
    k_utdt<<<ew(Bz * Uu), 256, 0, stream>>>(ut, utb, dtp, Bz * Uu);
    gemm(utb, Bsb, nullptr, zcur, out, Uu, Dd, Dd, Dd, 4);          // z_next -> d_out
    k_cvt_f2b<<<ew(Bz * Dd), 256, 0, stream>>>(out, zb, Bz * Dd);   // z_next bf16
    float* yt = out + (size_t)Bz * Dd;
    gemm(zb,  Cb, nullptr, nullptr, yt, Dd, NPAD, NOBS, NOBS, 1);   // C term
    gemm(utb, Db, nullptr, yt,      yt, Uu, NPAD, NOBS, NOBS, 1);   // + D term
}
